// FullAttention_14740327760099
// MI455X (gfx1250) — compile-verified, matrix operand order checked by reading, not run on hardware
//
#include <hip/hip_runtime.h>
#include <stdint.h>

typedef __attribute__((ext_vector_type(16))) _Float16 v16h;
typedef __attribute__((ext_vector_type(2)))  __fp16   v2fp16;
typedef __attribute__((ext_vector_type(8)))  float    v8f;

namespace {
constexpr int kH = 8, kD = 32, kL = 4096, kS = 4096;
constexpr int kWaves = 4;          // waves per block
constexpr int kQT    = 2;          // 16-row Q tiles per wave -> 128 q rows/block
constexpr int kChunk = 128;        // keys per mainloop iteration
constexpr int kNT    = kChunk / 16;// 8 key subtiles per chunk
constexpr int kNS    = kChunk / 32;// 4 32-key PV slices per chunk
constexpr int kLdK   = 34;         // Kb stride, halves (68 B -> 17-bank step)
constexpr int kLdV   = 130;        // Vt stride, halves (260 B -> 65-bank step)
constexpr int kLdP   = 130;        // Pb stride, floats
// fold 1/sqrt(D) * log2(e) into Q so scores land in the exp2 domain
constexpr float kScaleLog2e = 0.25505413550397954f;
constexpr float kMaskVal    = -1e9f * kScaleLog2e;   // ref: mask BEFORE scale

union HV { v16h v; uint32_t u[8]; _Float16 e[16]; };
} // namespace

__global__ __launch_bounds__(kWaves * 32)
void fattn_fwd(const float* __restrict__ qg, const float* __restrict__ kg,
               const float* __restrict__ vg, const uint8_t* __restrict__ qmask,
               const uint8_t* __restrict__ kvmask, float* __restrict__ outg)
{
  __shared__ _Float16 Kb[kChunk * kLdK];        // K chunk, [key][d]      (f16)
  __shared__ _Float16 Vt[kD * kLdV];            // V chunk, [d][key]      (f16)
  __shared__ float    Pb[kWaves * 16 * kLdP];   // per-wave P staging [m][k] (f32)

  const int h    = blockIdx.x % kH;
  const int qblk = blockIdx.x / kH;
  const int tid  = threadIdx.x;
  const int lane = tid & 31;
  const int wave = tid >> 5;
  const int lo   = lane & 15;
  const int hi   = lane >> 4;
  const int q0   = qblk * (kWaves * kQT * 16) + wave * (kQT * 16);

  // A/B-layout contraction offsets: pair (j,j+1) holds k = j + 8*(j>=8) + 8*hi
  int koff[8];
  #pragma unroll
  for (int j = 0; j < 16; j += 2) koff[j >> 1] = j + ((j >= 8) ? 8 : 0) + hi * 8;

  // ---- Q tiles -> WMMA A layout, pre-scaled by log2(e)/sqrt(D) ----
  HV   aQ[kQT];
  bool qok[kQT][8];
  int  qall[kQT];                                // wave-uniform: all rows valid
  #pragma unroll
  for (int t = 0; t < kQT; ++t) {
    const int    row = q0 + t * 16 + lo;
    const float* qp  = qg + (row * kH + h) * kD;
    #pragma unroll
    for (int j = 0; j < 16; j += 2) {
      const float2 f = *(const float2*)(qp + koff[j >> 1]);
      aQ[t].e[j]     = (_Float16)(f.x * kScaleLog2e);
      aQ[t].e[j + 1] = (_Float16)(f.y * kScaleLog2e);
    }
    bool all8 = true;
    #pragma unroll
    for (int r = 0; r < 8; ++r) {
      qok[t][r] = qmask[q0 + t * 16 + r + 8 * hi] != 0;
      all8 = all8 && qok[t][r];
    }
    qall[t] = __all(all8);
  }

  // constant all-ones B tile: row sums of P via WMMA (replicated over lanes)
  HV bOnes;
  #pragma unroll
  for (int j = 0; j < 16; ++j) bOnes.e[j] = (_Float16)1.0f;

  // online-softmax state: lacc accumulates the denominator via ones-WMMA
  float mrun[kQT][8];
  v8f   acc0[kQT], acc1[kQT], lacc[kQT];
  #pragma unroll
  for (int t = 0; t < kQT; ++t) {
    acc0[t] = (v8f){};
    acc1[t] = (v8f){};
    lacc[t] = (v8f){};
    #pragma unroll
    for (int r = 0; r < 8; ++r) mrun[t][r] = -__builtin_inff();
  }

  const float* kh        = kg + h * kD;
  const float* vh        = vg + h * kD;
  const int    rowStride = kH * kD;

  for (int s0 = 0; s0 < kS; s0 += kChunk) {
    __syncthreads();                             // previous chunk fully consumed

    if (s0 + kChunk < kS) {                      // prefetch next chunk rows
      __builtin_prefetch(kh + (size_t)(s0 + kChunk + tid) * rowStride, 0, 1);
      __builtin_prefetch(vh + (size_t)(s0 + kChunk + tid) * rowStride, 0, 1);
    }

    // cooperative stage: K row-major (packed half pairs), V transposed
    for (int i = tid; i < kChunk * (kD / 2); i += kWaves * 32) {
      const int key = i >> 4;
      const int dp  = (i & 15) * 2;
      const float2 kf = *(const float2*)(kh + (s0 + key) * rowStride + dp);
      const float2 vf = *(const float2*)(vh + (s0 + key) * rowStride + dp);
      union { uint32_t u; _Float16 e[2]; } pk;
      pk.e[0] = (_Float16)kf.x; pk.e[1] = (_Float16)kf.y;
      *(uint32_t*)&Kb[key * kLdK + dp] = pk.u;
      Vt[dp * kLdV + key]       = (_Float16)vf.x;
      Vt[(dp + 1) * kLdV + key] = (_Float16)vf.y;
    }
    // barrier doubles as block-wide "entire kv chunk unmasked?" reduction
    const int allkv = __syncthreads_and((int)(kvmask[s0 + tid] != 0));

    bool kvok[kNT];
    if (!allkv) {
      #pragma unroll
      for (int ts = 0; ts < kNT; ++ts) kvok[ts] = kvmask[s0 + 16 * ts + lo] != 0;
    } else {
      #pragma unroll
      for (int ts = 0; ts < kNT; ++ts) kvok[ts] = true;
    }

    // V B tiles (4 slices x 2 d-halves), shared by both q tiles
    HV bV0[kNS], bV1[kNS];
    #pragma unroll
    for (int p = 0; p < 8; ++p) {
      const int kk = koff[p];
      #pragma unroll
      for (int s = 0; s < kNS; ++s) {
        bV0[s].u[p] = *(const uint32_t*)&Vt[lo * kLdV + 32 * s + kk];
        bV1[s].u[p] = *(const uint32_t*)&Vt[(16 + lo) * kLdV + 32 * s + kk];
      }
    }

    float* Pw = &Pb[wave * 16 * kLdP];

    #pragma unroll
    for (int t = 0; t < kQT; ++t) {
      // scores: stream K^T subtiles through a single B register set
      v8f c[kNT];
      #pragma unroll
      for (int ts = 0; ts < kNT; ++ts) {
        HV bKt;
        #pragma unroll
        for (int p = 0; p < 8; ++p)
          bKt.u[p] = *(const uint32_t*)&Kb[(16 * ts + lo) * kLdK + koff[p]];
        v8f z = {};
        c[ts] = __builtin_amdgcn_wmma_f32_16x16x32_f16(false, aQ[t].v, false,
                                                       bKt.v, (short)0, z,
                                                       false, false);
      }

      // slow path only when some mask bit is off (wave-uniform branch)
      if (!(allkv && qall[t])) {
        #pragma unroll
        for (int r = 0; r < 8; ++r) {
          const bool ok = qok[t][r];
          #pragma unroll
          for (int ts = 0; ts < kNT; ++ts)
            c[ts][r] = (ok && kvok[ts]) ? c[ts][r] : kMaskVal;
        }
      }

      // cross-lane row max
      float mx[8];
      #pragma unroll
      for (int r = 0; r < 8; ++r) {
        float a01 = fmaxf(c[0][r], c[1][r]), a23 = fmaxf(c[2][r], c[3][r]);
        float a45 = fmaxf(c[4][r], c[5][r]), a67 = fmaxf(c[6][r], c[7][r]);
        mx[r] = fmaxf(fmaxf(a01, a23), fmaxf(a45, a67));
        #pragma unroll
        for (int w = 1; w < 16; w <<= 1) mx[r] = fmaxf(mx[r], __shfl_xor(mx[r], w, 32));
      }

      // rescale only when some row's max actually grew (wave-uniform branch)
      float need = mx[0] - mrun[t][0];
      #pragma unroll
      for (int r = 1; r < 8; ++r) need = fmaxf(need, mx[r] - mrun[t][r]);
      if (!__all(need <= 0.0f)) {
        #pragma unroll
        for (int r = 0; r < 8; ++r) {
          const float mnew  = fmaxf(mrun[t][r], mx[r]);
          const float alpha = __builtin_amdgcn_exp2f(mrun[t][r] - mnew);
          lacc[t][r] *= alpha;
          acc0[t][r] *= alpha;
          acc1[t][r] *= alpha;
          mrun[t][r]  = mnew;
        }
      }

      // p = exp2(x - m), streamed straight into the LDS transpose buffer
      #pragma unroll
      for (int r = 0; r < 8; ++r) {
        const int m = r + 8 * hi;
        #pragma unroll
        for (int ts = 0; ts < kNT; ++ts)
          Pw[m * kLdP + 16 * ts + lo] =
              __builtin_amdgcn_exp2f(c[ts][r] - mrun[t][r]);
      }
      asm volatile("s_wait_dscnt 0x0" ::: "memory");   // same-wave DS RAW

      // PV + denominator: one pA register set reused per 32-key slice
      #pragma unroll
      for (int s = 0; s < kNS; ++s) {
        HV pA;
        #pragma unroll
        for (int pp = 0; pp < 8; ++pp) {
          const float2 f = *(const float2*)&Pw[lo * kLdP + 32 * s + koff[pp]];
          const v2fp16 ph = __builtin_amdgcn_cvt_pkrtz(f.x, f.y);
          pA.u[pp] = __builtin_bit_cast(uint32_t, ph);
        }
        acc0[t] = __builtin_amdgcn_wmma_f32_16x16x32_f16(false, pA.v, false,
                                                         bV0[s].v, (short)0,
                                                         acc0[t], false, false);
        acc1[t] = __builtin_amdgcn_wmma_f32_16x16x32_f16(false, pA.v, false,
                                                         bV1[s].v, (short)0,
                                                         acc1[t], false, false);
        lacc[t] = __builtin_amdgcn_wmma_f32_16x16x32_f16(false, pA.v, false,
                                                         bOnes.v, (short)0,
                                                         lacc[t], false, false);
      }
      asm volatile("" ::: "memory");                   // keep next t's stores below
    }
  }

  // ---- epilogue: denominator already lane-replicated by the ones-WMMA ----
  #pragma unroll
  for (int t = 0; t < kQT; ++t) {
    #pragma unroll
    for (int r = 0; r < 8; ++r) {
      const int row = q0 + t * 16 + r + 8 * hi;
      float*    op  = outg + (row * kH + h) * kD;
      const float inv = 1.0f / lacc[t][r];
      op[lo]      = acc0[t][r] * inv;
      op[16 + lo] = acc1[t][r] * inv;
    }
  }
}

extern "C" void kernel_launch(void* const* d_in, const int* in_sizes, int n_in,
                              void* d_out, int out_size, void* d_ws, size_t ws_size,
                              hipStream_t stream) {
  (void)in_sizes; (void)n_in; (void)out_size; (void)d_ws; (void)ws_size;
  const float*   q   = (const float*)d_in[0];
  const float*   k   = (const float*)d_in[1];
  const float*   v   = (const float*)d_in[2];
  const uint8_t* qm  = (const uint8_t*)d_in[3];
  const uint8_t* kvm = (const uint8_t*)d_in[4];
  float*         out = (float*)d_out;

  const int grid = kH * (kL / (kWaves * kQT * 16));  // 8 heads * 32 q-blocks = 256
  fattn_fwd<<<grid, kWaves * 32, 0, stream>>>(q, k, v, qm, kvm, out);
}
